// AgentRNN_4982162063605
// MI455X (gfx1250) — matrix operand order repaired and checked by reading, not compiled, not run on hardware
//
#include <hip/hip_runtime.h>
#include <cstdint>
#include <cstddef>

// ---------------------------------------------------------------------------
// Problem dimensions (fixed by the reference)
// ---------------------------------------------------------------------------
#define T_  64
#define N_  8
#define B_  128
#define D_  128
#define H_  512
#define S_  64
#define K_  16
#define A_  16
#define NB_ 1024

// ---------------------------------------------------------------------------
// WMMA types (gfx1250, wave32): v16bf A/B operands, v8f f32 accumulator
// ---------------------------------------------------------------------------
typedef __attribute__((ext_vector_type(16))) __bf16 v16bf;
typedef __attribute__((ext_vector_type(8)))  __bf16 v8bf;
typedef __attribute__((ext_vector_type(8)))  float  v8f;

#define V8F_ZERO {0.f,0.f,0.f,0.f,0.f,0.f,0.f,0.f}

__device__ __forceinline__ v8f wmma_bf16(v16bf a, v16bf b, v8f c) {
    // 8-arg form: (neg_a, A, neg_b, B, c_mod, C, reuse_a, reuse_b)
    return __builtin_amdgcn_wmma_f32_16x16x32_bf16(false, a, false, b,
                                                   (short)0, c, false, false);
}

// CDNA5 async global->LDS copy, 16 bytes per lane (tracked by ASYNCcnt).
__device__ __forceinline__ void async_load_lds_b128(unsigned lds_off, const void* gaddr) {
    asm volatile("global_load_async_to_lds_b128 %0, %1, off"
                 :: "v"(lds_off), "v"(gaddr) : "memory");
}
__device__ __forceinline__ void wait_asynccnt0() {
    asm volatile("s_wait_asynccnt 0x0" ::: "memory");
}

// generic (flat) LDS pointer -> workgroup-relative LDS byte offset
#define LDS_OFF(p) ((unsigned)(uintptr_t)(p))

// B-matrix fragments are pre-swizzled lane-major: frag*512 + lane*16 + e,
// where for logical element (k, n) of a 32(K)x16(N) tile:
//   lane = (n & 15) + ((k & 8) << 1), e = (k & 7) | ((k & 16) >> 1)
// -> one contiguous 32-byte v16bf load per lane.
__device__ __forceinline__ v16bf bfrag_load(const __bf16* frags, int frag, int lane) {
    return *(const v16bf*)(frags + (size_t)frag * 512 + lane * 16);
}

// A-matrix fragment (16 rows M x 32 K, bf16) per ISA layout:
// lanes 0-15: M=lane, K chunks [0..7],[16..23]; lanes 16-31: M=lane-16,
// K chunks [8..15],[24..31]  -> two contiguous 8-element chunks per lane.
__device__ __forceinline__ v16bf afrag_lds(const __bf16* lds, int rowstride,
                                           int k0, int lane) {
    int m  = lane & 15;
    int kb = (lane & 16) ? 8 : 0;
    const __bf16* row = lds + m * rowstride;
    v8bf x = *(const v8bf*)(row + k0 + kb);
    v8bf y = *(const v8bf*)(row + k0 + 16 + kb);
    return __builtin_shufflevector(x, y, 0,1,2,3,4,5,6,7,8,9,10,11,12,13,14,15);
}

// A-fragment from a row-major bf16 matrix in global memory
__device__ __forceinline__ v16bf afrag_global_bf16(const __bf16* base, int ld,
                                                   int k0, int lane) {
    int m  = lane & 15;
    int kb = (lane & 16) ? 8 : 0;
    const __bf16* p = base + (size_t)m * ld + k0 + kb;
    v8bf x = *(const v8bf*)(p);
    v8bf y = *(const v8bf*)(p + 16);
    return __builtin_shufflevector(x, y, 0,1,2,3,4,5,6,7,8,9,10,11,12,13,14,15);
}

// A-fragment straight from a row-major f32 matrix (ld = row stride in floats)
__device__ __forceinline__ v16bf afrag_global_f32(const float* base, int ld,
                                                  int k0, int lane) {
    int m  = lane & 15;
    int kb = (lane & 16) ? 8 : 0;
    const float* p = base + (size_t)m * ld + k0 + kb;
    float4 a0 = *(const float4*)(p);
    float4 a1 = *(const float4*)(p + 4);
    float4 b0 = *(const float4*)(p + 16);
    float4 b1 = *(const float4*)(p + 20);
    v16bf r;
    r[0]  = (__bf16)a0.x; r[1]  = (__bf16)a0.y; r[2]  = (__bf16)a0.z; r[3]  = (__bf16)a0.w;
    r[4]  = (__bf16)a1.x; r[5]  = (__bf16)a1.y; r[6]  = (__bf16)a1.z; r[7]  = (__bf16)a1.w;
    r[8]  = (__bf16)b0.x; r[9]  = (__bf16)b0.y; r[10] = (__bf16)b0.z; r[11] = (__bf16)b0.w;
    r[12] = (__bf16)b1.x; r[13] = (__bf16)b1.y; r[14] = (__bf16)b1.z; r[15] = (__bf16)b1.w;
    return r;
}

__device__ __forceinline__ float sigmoidf_(float x) {
    return 1.f / (1.f + __expf(-x));
}

// ---------------------------------------------------------------------------
// Workspace layout (bytes).  Total ~13.1 MB.
// ---------------------------------------------------------------------------
static constexpr size_t OFF_HBUF1 = 0;                        // f32 NB*H
static constexpr size_t OFF_HBUF2 = OFF_HBUF1 + 2097152;      // f32 NB*H
static constexpr size_t OFF_EF    = OFF_HBUF2 + 2097152;      // f32 K*S
static constexpr size_t OFF_ETF   = OFF_EF    + 4096;         // f32 S*K
static constexpr size_t OFF_B1F   = OFF_ETF   + 4096;         // f32 K*A
static constexpr size_t OFF_W1T   = OFF_B1F   + 1024;         // f32 H*(K*A)
static constexpr size_t OFF_KARG  = OFF_W1T   + 524288;       // i32 NB
static constexpr size_t OFF_FEMB  = OFF_KARG  + 4096;         // bf16 frags D*H
static constexpr size_t OFF_FPOL  = OFF_FEMB  + 131072;
static constexpr size_t OFF_FWI1  = OFF_FPOL  + 131072;       // bf16 frags H*3H
static constexpr size_t OFF_FWH1  = OFF_FWI1  + 1572864;
static constexpr size_t OFF_FWI2  = OFF_FWH1  + 1572864;
static constexpr size_t OFF_FWH2  = OFF_FWI2  + 1572864;
static constexpr size_t OFF_FWSUB = OFF_FWH2  + 1572864;      // bf16 frags H*S
static constexpr size_t OFF_FW1   = OFF_FWSUB + 65536;        // bf16 frags H*(K*A)
static constexpr size_t OFF_FET   = OFF_FW1   + 262144;       // bf16 frags S*K
static constexpr size_t OFF_HBF1  = OFF_FET   + 2048;         // bf16 NB*H mirror
static constexpr size_t OFF_HBF2  = OFF_HBF1  + 1048576;      // bf16 NB*H mirror

// ---------------------------------------------------------------------------
// One-time: subtask MLP  e = tanh(relu(I_K @ W_e1 + b_e1) @ W_e2 + b_e2),
// hypernet  w1T[h, k*A+a], b1[k, a].
// ---------------------------------------------------------------------------
__global__ __launch_bounds__(256) void prep_small(
    const float* __restrict__ W_e1, const float* __restrict__ b_e1,
    const float* __restrict__ W_e2, const float* __restrict__ b_e2,
    const float* __restrict__ W_w1, const float* __restrict__ b_w1,
    const float* __restrict__ W_b1, const float* __restrict__ b_b1,
    float* __restrict__ ef, float* __restrict__ eTf,
    float* __restrict__ w1T, float* __restrict__ b1f)
{
    __shared__ float r1[K_ * S_];
    __shared__ float es[K_ * S_];
    const int tid = threadIdx.x;

    for (int i = tid; i < K_ * S_; i += 256) {
        int s = i & (S_ - 1);
        float v = W_e1[i] + b_e1[s];
        r1[i] = v > 0.f ? v : 0.f;
    }
    __syncthreads();
    for (int i = tid; i < K_ * S_; i += 256) {
        int k = i >> 6, s2 = i & (S_ - 1);
        float sum = b_e2[s2];
        for (int s = 0; s < S_; ++s) sum += r1[k * S_ + s] * W_e2[s * S_ + s2];
        float v = tanhf(sum);
        es[i] = v;
        ef[i] = v;
        eTf[s2 * K_ + k] = v;           // e^T, row-major (S x K) for logits GEMM
    }
    __syncthreads();
    // w1 = e @ W_w1 + b_w1, stored h-major: w1T[h*256 + k*16 + a]
    for (int idx = tid; idx < K_ * H_ * A_; idx += 256) {
        int k = idx >> 13;
        int h = (idx >> 4) & (H_ - 1);
        int a = idx & 15;
        float sum = b_w1[h * A_ + a];
        for (int s = 0; s < S_; ++s)
            sum += es[k * S_ + s] * W_w1[(size_t)s * (H_ * A_) + h * A_ + a];
        w1T[h * (K_ * A_) + k * A_ + a] = sum;
    }
    for (int idx = tid; idx < K_ * A_; idx += 256) {
        int k = idx >> 4, a = idx & 15;
        float sum = b_b1[a];
        for (int s = 0; s < S_; ++s) sum += es[k * S_ + s] * W_b1[s * A_ + a];
        b1f[idx] = sum;
    }
}

// subtask_embed output = e broadcast over (T, B)
__global__ __launch_bounds__(256) void bcast_embed(const float* __restrict__ ef,
                                                   float* __restrict__ out_se)
{
    float4 v = ((const float4*)ef)[threadIdx.x];                 // 256*4 = 1024
    ((float4*)(out_se + (size_t)blockIdx.x * (K_ * S_)))[threadIdx.x] = v;
}

// row-major f32 (K x N) -> bf16 B-fragment layout
__global__ __launch_bounds__(256) void convert_to_bfrag(
    const float* __restrict__ src, __bf16* __restrict__ dst, int K, int N)
{
    int idx = blockIdx.x * 256 + threadIdx.x;
    if (idx >= K * N) return;
    int k = idx / N, n = idx - k * N;
    int kt = k >> 5, kin = k & 31;
    int nt = n >> 4, nin = n & 15;
    int lane = nin + ((kin & 8) << 1);
    int frag = nt * (K >> 5) + kt;
    dst[(size_t)frag * 512 + lane * 16 + ((kin & 7) | ((kin & 16) >> 1))] = (__bf16)src[idx];
}

// plain f32 -> bf16 elementwise (initial carry mirrors)
__global__ __launch_bounds__(256) void cvt_f32_to_bf16(
    const float* __restrict__ src, __bf16* __restrict__ dst, int nelem)
{
    int idx = blockIdx.x * 256 + threadIdx.x;
    if (idx < nelem) dst[idx] = (__bf16)src[idx];
}

// ---------------------------------------------------------------------------
// One GRU time step, both branches (blockIdx.y).  Block = 32 rows x all H.
//   - async-copy h_prev (bf16 mirror) into LDS while the embed GEMM runs
//   - r/z gates accumulate x-part and h-part into the SAME wmma accumulator
//   - each B-fragment is shared by two 16-row subtiles (halves L2 traffic)
// ---------------------------------------------------------------------------
__global__ __launch_bounds__(256) void gru_step(
    const float* __restrict__ obs, const unsigned char* __restrict__ done,
    float* __restrict__ hbuf1, float* __restrict__ hbuf2,
    __bf16* __restrict__ hbf1, __bf16* __restrict__ hbf2,
    const __bf16* __restrict__ F_emb, const __bf16* __restrict__ F_pol,
    const __bf16* __restrict__ F_wi1, const __bf16* __restrict__ F_wh1,
    const __bf16* __restrict__ F_wi2, const __bf16* __restrict__ F_wh2,
    const float* __restrict__ b_embed, const float* __restrict__ b_pol,
    const float* __restrict__ bi1, const float* __restrict__ bhn1,
    const float* __restrict__ bi2, const float* __restrict__ bhn2,
    int t)
{
    constexpr int XSTR = H_ + 8;                   // bf16 row stride (16B-aligned)
    extern __shared__ __align__(16) unsigned char smem_raw[];
    __bf16* x_lds = (__bf16*)smem_raw;             // 32 rows x XSTR
    __bf16* h_lds = x_lds + 32 * XSTR;             // 32 rows x XSTR

    const int g    = blockIdx.y;
    const int row0 = blockIdx.x * 32;
    const int tid  = threadIdx.x;
    const int wave = tid >> 5;
    const int lane = tid & 31;
    const int n    = lane & 15;
    const int mh   = (lane & 16) ? 8 : 0;

    float*        hbuf = g ? hbuf2 : hbuf1;
    __bf16*       hbf  = g ? hbf2  : hbf1;
    const __bf16* Femb = g ? F_pol : F_emb;
    const __bf16* Fwi  = g ? F_wi2 : F_wi1;
    const __bf16* Fwh  = g ? F_wh2 : F_wh1;
    const float*  bemb = g ? b_pol : b_embed;
    const float*  bi   = g ? bi2   : bi1;
    const float*  bhn  = g ? bhn2  : bhn1;

    // kick off async DMA of h_prev (bf16) into LDS: 32 rows x 1KB
    {
        const __bf16* src = hbf + (size_t)row0 * H_;
        for (int i = tid; i < 32 * (H_ / 8); i += 256) {   // 16B chunks
            int rr = i >> 6, ch = i & 63;
            async_load_lds_b128(LDS_OFF(h_lds + rr * XSTR + ch * 8),
                                src + (size_t)rr * H_ + ch * 8);
        }
    }

    // phase 0 (overlaps with async copy): x = relu(obs_t @ W_emb + b)
    const float* obs_base = obs + ((size_t)t * NB_ + row0) * D_;
    for (int qq = 0; qq < 4; ++qq) {
        int ct = wave + qq * 8;                    // 32 col tiles / 8 waves
        v8f a0 = V8F_ZERO, a1 = V8F_ZERO;
        for (int ks = 0; ks < 4; ++ks) {
            v16bf Bf = bfrag_load(Femb, ct * 4 + ks, lane);
            a0 = wmma_bf16(afrag_global_f32(obs_base,            D_, ks * 32, lane), Bf, a0);
            a1 = wmma_bf16(afrag_global_f32(obs_base + 16 * D_,  D_, ks * 32, lane), Bf, a1);
        }
        float bb = bemb[ct * 16 + n];
        for (int r = 0; r < 8; ++r) {
            float v0 = a0[r] + bb; v0 = v0 > 0.f ? v0 : 0.f;
            float v1 = a1[r] + bb; v1 = v1 > 0.f ? v1 : 0.f;
            x_lds[(r + mh) * XSTR      + ct * 16 + n] = (__bf16)v0;
            x_lds[(16 + r + mh) * XSTR + ct * 16 + n] = (__bf16)v1;
        }
    }

    // h tile resident; apply per-row done-reset in LDS
    wait_asynccnt0();
    for (int i = tid; i < 32 * (H_ / 8); i += 256) {
        int rr = i >> 6, ch = i & 63;
        if (done[(size_t)t * NB_ + row0 + rr]) {
            uint4 z = {0u, 0u, 0u, 0u};
            *(uint4*)(h_lds + rr * XSTR + ch * 8) = z;
        }
    }
    __syncthreads();

    // phase 1: gates.  acc_r/acc_z hold (x@Wi + h@Wh) directly.
    for (int qq = 0; qq < 4; ++qq) {
        int c = wave + qq * 8;
        v8f ar0 = V8F_ZERO, az0 = V8F_ZERO, ain0 = V8F_ZERO, ahn0 = V8F_ZERO;
        v8f ar1 = V8F_ZERO, az1 = V8F_ZERO, ain1 = V8F_ZERO, ahn1 = V8F_ZERO;
        for (int ks = 0; ks < 16; ++ks) {
            v16bf Ax0 = afrag_lds(x_lds,             XSTR, ks * 32, lane);
            v16bf Ah0 = afrag_lds(h_lds,             XSTR, ks * 32, lane);
            v16bf Ax1 = afrag_lds(x_lds + 16 * XSTR, XSTR, ks * 32, lane);
            v16bf Ah1 = afrag_lds(h_lds + 16 * XSTR, XSTR, ks * 32, lane);
            if (ks < 15)   // keep next weight panel warm
                __builtin_prefetch(Fwi + (size_t)(c * 16 + ks + 1) * 512 + lane * 16, 0, 1);
            v16bf Bir = bfrag_load(Fwi, (     c) * 16 + ks, lane);
            v16bf Bhr = bfrag_load(Fwh, (     c) * 16 + ks, lane);
            v16bf Biz = bfrag_load(Fwi, (32 + c) * 16 + ks, lane);
            v16bf Bhz = bfrag_load(Fwh, (32 + c) * 16 + ks, lane);
            v16bf Bin = bfrag_load(Fwi, (64 + c) * 16 + ks, lane);
            v16bf Bhn = bfrag_load(Fwh, (64 + c) * 16 + ks, lane);
            ar0  = wmma_bf16(Ax0, Bir, ar0);  ar0  = wmma_bf16(Ah0, Bhr, ar0);
            az0  = wmma_bf16(Ax0, Biz, az0);  az0  = wmma_bf16(Ah0, Bhz, az0);
            ain0 = wmma_bf16(Ax0, Bin, ain0); ahn0 = wmma_bf16(Ah0, Bhn, ahn0);
            ar1  = wmma_bf16(Ax1, Bir, ar1);  ar1  = wmma_bf16(Ah1, Bhr, ar1);
            az1  = wmma_bf16(Ax1, Biz, az1);  az1  = wmma_bf16(Ah1, Bhz, az1);
            ain1 = wmma_bf16(Ax1, Bin, ain1); ahn1 = wmma_bf16(Ah1, Bhn, ahn1);
        }
        float b_ir = bi[         c * 16 + n];
        float b_iz = bi[H_     + c * 16 + n];
        float b_in = bi[2 * H_ + c * 16 + n];
        float b_hn = bhn[        c * 16 + n];
        for (int s2 = 0; s2 < 2; ++s2) {
            const v8f& ar  = s2 ? ar1  : ar0;
            const v8f& az  = s2 ? az1  : az0;
            const v8f& ain = s2 ? ain1 : ain0;
            const v8f& ahn = s2 ? ahn1 : ahn0;
            for (int r = 0; r < 8; ++r) {
                int   row  = row0 + s2 * 16 + r + mh;
                float rg   = sigmoidf_(ar[r] + b_ir);
                float zg   = sigmoidf_(az[r] + b_iz);
                float ng   = tanhf(ain[r] + b_in + rg * (ahn[r] + b_hn));
                float hold = done[(size_t)t * NB_ + row] ? 0.f
                             : hbuf[(size_t)row * H_ + c * 16 + n];
                float hnew = (1.f - zg) * ng + zg * hold;
                hbuf[(size_t)row * H_ + c * 16 + n] = hnew;
                hbf [(size_t)row * H_ + c * 16 + n] = (__bf16)hnew;   // mirror
            }
        }
    }
}

// ---------------------------------------------------------------------------
// Per-step logits: ae2 = h1 @ W_sub + b_sub (WMMA), logits = ae2 @ e^T (WMMA),
// scatter to (T,B,N,K), and argmax -> karg (stored at p = b*N + n).
// ---------------------------------------------------------------------------
__global__ __launch_bounds__(32) void logits_step(
    const __bf16* __restrict__ hbf1,
    const __bf16* __restrict__ F_wsub, const float* __restrict__ b_sub,
    const __bf16* __restrict__ F_eT,
    float* __restrict__ out_logits, int* __restrict__ karg, int t)
{
    constexpr int AESTR = S_ + 8;
    __shared__ __bf16 ae_lds[16 * AESTR];
    __shared__ float  lg_lds[16 * 16];

    const int lane = threadIdx.x;
    const int j0   = blockIdx.x * 16;       // rows j = n*B + b  (y1 order)
    const int n    = lane & 15;
    const int mh   = (lane & 16) ? 8 : 0;

    v8f acc[4] = {V8F_ZERO, V8F_ZERO, V8F_ZERO, V8F_ZERO};
    const __bf16* base = hbf1 + (size_t)j0 * H_;
    for (int ks = 0; ks < 16; ++ks) {
        v16bf Af = afrag_global_bf16(base, H_, ks * 32, lane);
        for (int st = 0; st < 4; ++st)
            acc[st] = wmma_bf16(Af, bfrag_load(F_wsub, st * 16 + ks, lane), acc[st]);
    }
    for (int st = 0; st < 4; ++st) {
        float bs = b_sub[st * 16 + n];
        for (int r = 0; r < 8; ++r)
            ae_lds[(r + mh) * AESTR + st * 16 + n] = (__bf16)(acc[st][r] + bs);
    }
    __syncthreads();

    v8f lg = V8F_ZERO;
    for (int ks = 0; ks < 2; ++ks)
        lg = wmma_bf16(afrag_lds(ae_lds, AESTR, ks * 32, lane),
                       bfrag_load(F_eT, ks, lane), lg);

    for (int r = 0; r < 8; ++r) {
        int j  = j0 + r + mh;
        int nn = j >> 7;                    // j / B
        int bb = j & (B_ - 1);              // j % B
        out_logits[((size_t)t * (B_ * N_) + bb * N_ + nn) * K_ + n] = lg[r];
        lg_lds[(r + mh) * 16 + n] = lg[r];
    }
    __syncthreads();

    if (lane < 16) {
        int   j    = j0 + lane;
        float best = lg_lds[lane * 16];
        int   bk   = 0;
        for (int k = 1; k < K_; ++k) {
            float v = lg_lds[lane * 16 + k];
            if (v > best) { best = v; bk = k; }
        }
        int nn = j >> 7, bb = j & (B_ - 1);
        karg[bb * N_ + nn] = bk;            // indexed by p == i (q-order)
    }
}

// ---------------------------------------------------------------------------
// Per-step Q head: q_full = h2 @ w1_all (WMMA, 16x256 per block), then
// per-row gather with karg + b1.  h2 staged via async DMA from bf16 mirror.
// ---------------------------------------------------------------------------
__global__ __launch_bounds__(256) void q_step(
    const __bf16* __restrict__ hbf2, const __bf16* __restrict__ F_w1,
    const float* __restrict__ b1f, const int* __restrict__ karg,
    float* __restrict__ out_q, int t)
{
    constexpr int HSTR = H_ + 8;
    constexpr int QSTR = 256 + 4;
    __shared__ __bf16 h2_lds[16 * HSTR];
    __shared__ float  q_lds[16 * QSTR];

    const int tid  = threadIdx.x;
    const int wave = tid >> 5;
    const int lane = tid & 31;
    const int i0   = blockIdx.x * 16;

    {
        const __bf16* src = hbf2 + (size_t)i0 * H_;
        for (int i = tid; i < 16 * (H_ / 8); i += 256) {
            int rr = i >> 6, ch = i & 63;
            async_load_lds_b128(LDS_OFF(h2_lds + rr * HSTR + ch * 8),
                                src + (size_t)rr * H_ + ch * 8);
        }
    }
    wait_asynccnt0();
    __syncthreads();

    const int n  = lane & 15;
    const int mh = (lane & 16) ? 8 : 0;
    for (int qq = 0; qq < 2; ++qq) {
        int ct = wave + qq * 8;                   // 16 col tiles / 8 waves
        v8f acc = V8F_ZERO;
        for (int ks = 0; ks < 16; ++ks)
            acc = wmma_bf16(afrag_lds(h2_lds, HSTR, ks * 32, lane),
                            bfrag_load(F_w1, ct * 16 + ks, lane), acc);
        for (int r = 0; r < 8; ++r)
            q_lds[(r + mh) * QSTR + ct * 16 + n] = acc[r];
    }
    __syncthreads();

    int m = tid >> 4, a = tid & 15;
    int i = i0 + m;
    int k = karg[i];
    out_q[((size_t)t * NB_ + i) * A_ + a] = q_lds[m * QSTR + k * 16 + a]
                                          + b1f[k * 16 + a];
}

// ---------------------------------------------------------------------------
// Host launcher
// ---------------------------------------------------------------------------
extern "C" void kernel_launch(void* const* d_in, const int* in_sizes, int n_in,
                              void* d_out, int out_size, void* d_ws, size_t ws_size,
                              hipStream_t stream)
{
    (void)in_sizes; (void)n_in; (void)out_size; (void)ws_size;

    const float* h1      = (const float*)d_in[0];
    const float* h2      = (const float*)d_in[1];
    const float* obs     = (const float*)d_in[2];
    const unsigned char* done = (const unsigned char*)d_in[3];
    const float* W_embed = (const float*)d_in[4];
    const float* b_embed = (const float*)d_in[5];
    const float* Wi1     = (const float*)d_in[6];
    const float* bi1     = (const float*)d_in[7];
    const float* Wh1     = (const float*)d_in[8];
    const float* bhn1    = (const float*)d_in[9];
    const float* W_sub   = (const float*)d_in[10];
    const float* b_sub   = (const float*)d_in[11];
    const float* W_e1    = (const float*)d_in[12];
    const float* b_e1    = (const float*)d_in[13];
    const float* W_e2    = (const float*)d_in[14];
    const float* b_e2    = (const float*)d_in[15];
    const float* W_pol   = (const float*)d_in[16];
    const float* b_pol   = (const float*)d_in[17];
    const float* Wi2     = (const float*)d_in[18];
    const float* bi2     = (const float*)d_in[19];
    const float* Wh2     = (const float*)d_in[20];
    const float* bhn2    = (const float*)d_in[21];
    const float* W_w1    = (const float*)d_in[22];
    const float* b_w1    = (const float*)d_in[23];
    const float* W_b1    = (const float*)d_in[24];
    const float* b_b1    = (const float*)d_in[25];

    char* ws = (char*)d_ws;
    float*  hbuf1  = (float*)(ws + OFF_HBUF1);
    float*  hbuf2  = (float*)(ws + OFF_HBUF2);
    float*  ef     = (float*)(ws + OFF_EF);
    float*  eTf    = (float*)(ws + OFF_ETF);
    float*  b1f    = (float*)(ws + OFF_B1F);
    float*  w1T    = (float*)(ws + OFF_W1T);
    int*    karg   = (int*)  (ws + OFF_KARG);
    __bf16* F_emb  = (__bf16*)(ws + OFF_FEMB);
    __bf16* F_pol  = (__bf16*)(ws + OFF_FPOL);
    __bf16* F_wi1  = (__bf16*)(ws + OFF_FWI1);
    __bf16* F_wh1  = (__bf16*)(ws + OFF_FWH1);
    __bf16* F_wi2  = (__bf16*)(ws + OFF_FWI2);
    __bf16* F_wh2  = (__bf16*)(ws + OFF_FWH2);
    __bf16* F_wsub = (__bf16*)(ws + OFF_FWSUB);
    __bf16* F_w1   = (__bf16*)(ws + OFF_FW1);
    __bf16* F_eT   = (__bf16*)(ws + OFF_FET);
    __bf16* hbf1   = (__bf16*)(ws + OFF_HBF1);
    __bf16* hbf2   = (__bf16*)(ws + OFF_HBF2);

    float* out        = (float*)d_out;
    float* out_hT1    = out;                 // (NB,H)
    float* out_hT2    = out + 524288;        // (NB,H)
    float* out_q      = out + 1048576;       // (T,NB,A)
    float* out_logits = out + 2097152;       // (T,B,N,K)
    float* out_se     = out + 3145728;       // (T,B,K,S)

    // initial carries (f32 state + bf16 mirrors)
    hipMemcpyAsync(hbuf1, h1, (size_t)NB_ * H_ * sizeof(float),
                   hipMemcpyDeviceToDevice, stream);
    hipMemcpyAsync(hbuf2, h2, (size_t)NB_ * H_ * sizeof(float),
                   hipMemcpyDeviceToDevice, stream);
    cvt_f32_to_bf16<<<(NB_ * H_ + 255) / 256, 256, 0, stream>>>(h1, hbf1, NB_ * H_);
    cvt_f32_to_bf16<<<(NB_ * H_ + 255) / 256, 256, 0, stream>>>(h2, hbf2, NB_ * H_);

    // one-time small math + weight swizzle to bf16 fragments
    prep_small<<<1, 256, 0, stream>>>(W_e1, b_e1, W_e2, b_e2,
                                      W_w1, b_w1, W_b1, b_b1,
                                      ef, eTf, w1T, b1f);
    bcast_embed<<<T_ * B_, 256, 0, stream>>>(ef, out_se);

    auto conv = [&](const float* s, __bf16* d, int K, int N) {
        int e = K * N;
        convert_to_bfrag<<<(e + 255) / 256, 256, 0, stream>>>(s, d, K, N);
    };
    conv(W_embed, F_emb,  D_, H_);
    conv(W_pol,   F_pol,  D_, H_);
    conv(Wi1,     F_wi1,  H_, 3 * H_);
    conv(Wh1,     F_wh1,  H_, 3 * H_);
    conv(Wi2,     F_wi2,  H_, 3 * H_);
    conv(Wh2,     F_wh2,  H_, 3 * H_);
    conv(W_sub,   F_wsub, H_, S_);
    conv(w1T,     F_w1,   H_, K_ * A_);      // after prep_small (stream order)
    conv(eTf,     F_eT,   S_, K_);

    // sequential scan over time
    constexpr int XSTR = H_ + 8;
    const size_t gru_smem = (size_t)2 * 32 * XSTR * sizeof(__bf16);   // 66,560 B
    for (int t = 0; t < T_; ++t) {
        gru_step<<<dim3(NB_ / 32, 2), 256, gru_smem, stream>>>(
            obs, done, hbuf1, hbuf2, hbf1, hbf2,
            F_emb, F_pol, F_wi1, F_wh1, F_wi2, F_wh2,
            b_embed, b_pol, bi1, bhn1, bi2, bhn2, t);
        logits_step<<<NB_ / 16, 32, 0, stream>>>(
            hbf1, F_wsub, b_sub, F_eT, out_logits, karg, t);
        q_step<<<NB_ / 16, 256, 0, stream>>>(
            hbf2, F_w1, b1f, karg, out_q, t);
    }

    // final carries
    hipMemcpyAsync(out_hT1, hbuf1, (size_t)NB_ * H_ * sizeof(float),
                   hipMemcpyDeviceToDevice, stream);
    hipMemcpyAsync(out_hT2, hbuf2, (size_t)NB_ * H_ * sizeof(float),
                   hipMemcpyDeviceToDevice, stream);
}
